// Net_90744069030475
// MI455X (gfx1250) — matrix operand order repaired and checked by reading, not compiled, not run on hardware
//
#include <hip/hip_runtime.h>

#define NN   4096
#define FF   32
#define HH   2
#define NHID 64
#define CC   10
#define JJ   3
#define EE   65536
#define GG   13
#define EN   (EE + NN)      // edges + self loops = 69632
#define KT   128            // 4096 / 32 k-tiles
#define CT   26             // 13 RHS * 2 col-tiles in the fragment buffer
#define RT   (NN / 16)      // 256 row tiles

typedef __attribute__((ext_vector_type(16))) __bf16 v16bf;
typedef __attribute__((ext_vector_type(8)))  float  v8f;

union Frag16 { v16bf v; uint4 q[2]; };

__device__ __forceinline__ unsigned short f2bfbits(float f) {
  union { float f; unsigned u; } a; a.f = f;
  unsigned r = a.u + 0x7FFFu + ((a.u >> 16) & 1u);   // RNE
  return (unsigned short)(r >> 16);
}
__device__ __forceinline__ __bf16 f2bf16(float f) {
  union { unsigned short s; __bf16 b; } o; o.s = f2bfbits(f);
  return o.b;
}

// ---------------------------------------------------------------------------
// Pack B operand [K=4096 rows, 32 cols] (fp32, row-major, ld=srcLd, col base
// srcColBase) into WMMA bf16 B-fragments at col-tiles [ctDst, ctDst+1].
// B layout (16-bit, 32x16): lane n in 0..15 holds col n, K = e; lanes 16..31
// hold col n, K = 16+e.
// ---------------------------------------------------------------------------
__global__ void k_pack(const float* __restrict__ src, int srcLd, int srcColBase,
                       unsigned short* __restrict__ frag, int ctDst, int doAbs) {
  int t = blockIdx.x * blockDim.x + threadIdx.x;
  if (t >= KT * 2 * 32) return;
  int lane = t & 31, ct = (t >> 5) & 1, kt = t >> 6;
  int col = srcColBase + ct * 16 + (lane & 15);
  int kb  = (kt << 5) + ((lane >> 4) << 4);
  unsigned short* dst = frag + ((size_t)(kt * CT + ctDst + ct) * 32 + lane) * 16;
  #pragma unroll
  for (int e = 0; e < 16; ++e) {
    float v = src[(size_t)(kb + e) * srcLd + col];
    if (doAbs) v = fabsf(v);
    dst[e] = f2bfbits(v);
  }
}

// ---------------------------------------------------------------------------
// OUT[16x32 per wave] = A[4096x4096] @ Bfrag(bf16), fp32 accum.
// A source: either fp32 row-major (converted to bf16 inline; optionally the
// converted fragments are saved to Asave) or pre-converted bf16 fragments
// (Afrag non-null) written by a previous pass.
// A-frag (16-bit 16x32): lane l<16 holds row l, K {0..15}; lane l+16
// holds row l, K {16..31}.
// ---------------------------------------------------------------------------
__global__ __launch_bounds__(256) void k_gemm(
    const float* __restrict__ A, long long Astride,
    const unsigned short* __restrict__ Bfrag, int ctBase,
    const unsigned short* __restrict__ Afrag,
    unsigned short* __restrict__ Asave,
    float* __restrict__ Out, int ldOut, int outZStride, int absOut) {
  const int lane = threadIdx.x & 31;
  const int wave = threadIdx.x >> 5;
  const int half = lane >> 4;
  const int l15  = lane & 15;
  const int rt   = blockIdx.x * 8 + wave;
  const int row0 = rt * 16;
  const int ctp  = blockIdx.y;
  const int z    = blockIdx.z;

  const unsigned short* Bbase =
      Bfrag + ((size_t)(ctBase + 2 * ctp) * 32 + lane) * 16;
  const size_t afIdx =
      (((size_t)z * RT + rt) * KT) * 512 + (size_t)lane * 16;

  v8f acc0 = {};
  v8f acc1 = {};

  if (Afrag) {
    // A already bf16 fragments (level-2 pass): half the bytes, no conversion.
    const unsigned short* Ap = Afrag + afIdx;
    #pragma unroll 2
    for (int kt = 0; kt < KT; ++kt) {
      Frag16 a;
      const unsigned short* ak = Ap + (size_t)kt * 512;
      a.q[0] = ((const uint4*)ak)[0];
      a.q[1] = ((const uint4*)ak)[1];
      const unsigned short* bk = Bbase + (size_t)kt * (CT * 512);
      Frag16 b0, b1;
      b0.q[0] = ((const uint4*)bk)[0];
      b0.q[1] = ((const uint4*)bk)[1];
      b1.q[0] = ((const uint4*)(bk + 512))[0];
      b1.q[1] = ((const uint4*)(bk + 512))[1];
      acc0 = __builtin_amdgcn_wmma_f32_16x16x32_bf16(false, a.v, false, b0.v,
                                                     (short)0, acc0, false, false);
      acc1 = __builtin_amdgcn_wmma_f32_16x16x32_bf16(false, a.v, false, b1.v,
                                                     (short)0, acc1, false, false);
    }
  } else {
    const float* Arow = A + (long long)z * Astride + (long long)(row0 + l15) * NN;
    #pragma unroll 2
    for (int kt = 0; kt < KT; ++kt) {
      const int k = kt << 5;
      const float4* ap0 = (const float4*)(Arow + k + half * 16);
      const float4* ap1 = (const float4*)(Arow + k + 8 + half * 16);
      float4 x0 = ap0[0], x1 = ap0[1], x2 = ap1[0], x3 = ap1[1];
      if (kt + 2 < KT) __builtin_prefetch(Arow + k + 64, 0, 1);

      Frag16 a;
      a.v[0]  = f2bf16(x0.x); a.v[1]  = f2bf16(x0.y);
      a.v[2]  = f2bf16(x0.z); a.v[3]  = f2bf16(x0.w);
      a.v[4]  = f2bf16(x1.x); a.v[5]  = f2bf16(x1.y);
      a.v[6]  = f2bf16(x1.z); a.v[7]  = f2bf16(x1.w);
      a.v[8]  = f2bf16(x2.x); a.v[9]  = f2bf16(x2.y);
      a.v[10] = f2bf16(x2.z); a.v[11] = f2bf16(x2.w);
      a.v[12] = f2bf16(x3.x); a.v[13] = f2bf16(x3.y);
      a.v[14] = f2bf16(x3.z); a.v[15] = f2bf16(x3.w);

      if (Asave) {  // spill converted psi fragments for the level-2 pass
        uint4* sv = (uint4*)(Asave + afIdx + (size_t)kt * 512);
        sv[0] = a.q[0];
        sv[1] = a.q[1];
      }

      const unsigned short* bk = Bbase + (size_t)kt * (CT * 512);
      Frag16 b0, b1;
      b0.q[0] = ((const uint4*)bk)[0];
      b0.q[1] = ((const uint4*)bk)[1];
      b1.q[0] = ((const uint4*)(bk + 512))[0];
      b1.q[1] = ((const uint4*)(bk + 512))[1];
      acc0 = __builtin_amdgcn_wmma_f32_16x16x32_bf16(false, a.v, false, b0.v,
                                                     (short)0, acc0, false, false);
      acc1 = __builtin_amdgcn_wmma_f32_16x16x32_bf16(false, a.v, false, b1.v,
                                                     (short)0, acc1, false, false);
    }
  }

  // D layout: VGPR r, lanes 0-15 -> (M=r, N=lane); lanes 16-31 -> (M=8+r).
  const int colBase = z * outZStride + ctp * 32;
  const int rr = row0 + half * 8;
  #pragma unroll
  for (int r = 0; r < 8; ++r) {
    float v0 = acc0[r], v1 = acc1[r];
    if (absOut) { v0 = fabsf(v0); v1 = fabsf(v1); }
    Out[(size_t)(rr + r) * ldOut + colBase + l15]      = v0;
    Out[(size_t)(rr + r) * ldOut + colBase + 16 + l15] = v1;
  }
}

// --------------------------- GAT branch kernels -----------------------------
__global__ void k_zero(unsigned* __restrict__ p, int n) {
  int i = blockIdx.x * blockDim.x + threadIdx.x;
  if (i < n) p[i] = 0u;
}

// h[g,n,o] = sum_f coef[n, g*32+f] * gat_W[g,f,o] + gat_b[g,o]
__global__ void k_h(const float* __restrict__ coef, const float* __restrict__ W,
                    const float* __restrict__ b, float* __restrict__ h) {
  int t = blockIdx.x * blockDim.x + threadIdx.x;
  int g = blockIdx.z;
  if (t >= NN * (HH * FF)) return;
  int n = t >> 6, o = t & 63;
  const float* c = coef + (size_t)n * (GG * FF) + g * FF;
  const float* w = W + ((size_t)g * FF) * (HH * FF) + o;
  float s = b[g * (HH * FF) + o];
  #pragma unroll 8
  for (int f = 0; f < FF; ++f) s += c[f] * w[(size_t)f * (HH * FF)];
  h[((size_t)g * NN + n) * (HH * FF) + o] = s;
}

// es/ed[g,n,hh] = sum_f h[g,n,hh*32+f] * att[g,hh,f]
__global__ void k_att(const float* __restrict__ h, const float* __restrict__ as_,
                      const float* __restrict__ ad_, float* __restrict__ es,
                      float* __restrict__ ed) {
  int t = blockIdx.x * blockDim.x + threadIdx.x;
  int g = blockIdx.z;
  if (t >= NN * HH) return;
  int n = t >> 1, hh = t & 1;
  const float* hp = h + ((size_t)g * NN + n) * (HH * FF) + hh * FF;
  const float* a0 = as_ + (g * HH + hh) * FF;
  const float* a1 = ad_ + (g * HH + hh) * FF;
  float s = 0.f, d = 0.f;
  #pragma unroll 8
  for (int f = 0; f < FF; ++f) { s += hp[f] * a0[f]; d += hp[f] * a1[f]; }
  es[((size_t)g * NN + n) * HH + hh] = s;
  ed[((size_t)g * NN + n) * HH + hh] = d;
}

__device__ __forceinline__ unsigned f2key(float f) {
  union { float f; unsigned u; } a; a.f = f;
  return (a.u & 0x80000000u) ? ~a.u : (a.u | 0x80000000u);
}
__device__ __forceinline__ float key2f(unsigned k) {
  union { unsigned u; float f; } a;
  a.u = (k & 0x80000000u) ? (k & 0x7FFFFFFFu) : ~k;
  return a.f;
}
__device__ __forceinline__ void edge_sd(const int* ei, int e, int& s, int& d) {
  if (e < EE) { s = ei[e]; d = ei[EE + e]; } else { s = e - EE; d = e - EE; }
}
__device__ __forceinline__ float leaky(float v) {
  return v > 0.f ? v : 0.2f * v;
}

__global__ void k_emax(const float* __restrict__ es, const float* __restrict__ ed,
                       const int* __restrict__ ei, unsigned* __restrict__ m) {
  int t = blockIdx.x * blockDim.x + threadIdx.x;
  int g = blockIdx.z;
  if (t >= EN * HH) return;
  int e = t >> 1, hh = t & 1, s, d;
  edge_sd(ei, e, s, d);
  float v = leaky(es[((size_t)g * NN + s) * HH + hh] +
                  ed[((size_t)g * NN + d) * HH + hh]);
  atomicMax(m + ((size_t)g * NN + d) * HH + hh, f2key(v));
}

__global__ void k_esum(const float* __restrict__ es, const float* __restrict__ ed,
                       const int* __restrict__ ei, const unsigned* __restrict__ m,
                       float* __restrict__ zs) {
  int t = blockIdx.x * blockDim.x + threadIdx.x;
  int g = blockIdx.z;
  if (t >= EN * HH) return;
  int e = t >> 1, hh = t & 1, s, d;
  edge_sd(ei, e, s, d);
  size_t di = ((size_t)g * NN + d) * HH + hh;
  float v = leaky(es[((size_t)g * NN + s) * HH + hh] + ed[di]);
  atomicAdd(zs + di, __expf(v - key2f(m[di])));
}

__global__ void k_eacc(const float* __restrict__ es, const float* __restrict__ ed,
                       const int* __restrict__ ei, const unsigned* __restrict__ m,
                       const float* __restrict__ zs, const float* __restrict__ h,
                       float* __restrict__ go) {
  int t = blockIdx.x * blockDim.x + threadIdx.x;
  int g = blockIdx.z;
  if (t >= EN * HH) return;
  int e = t >> 1, hh = t & 1, s, d;
  edge_sd(ei, e, s, d);
  size_t di = ((size_t)g * NN + d) * HH + hh;
  float v = leaky(es[((size_t)g * NN + s) * HH + hh] + ed[di]);
  float alpha = __expf(v - key2f(m[di])) / (zs[di] + 1e-16f);
  const float* hs = h + ((size_t)g * NN + s) * (HH * FF) + hh * FF;
  float* gd = go + ((size_t)g * NN + d) * (HH * FF) + hh * FF;
  #pragma unroll 8
  for (int f = 0; f < FF; ++f) atomicAdd(gd + f, alpha * hs[f]);
}

__device__ __forceinline__ float elu(float x) {
  return x > 0.f ? x : (__expf(x) - 1.f);
}

// yb[n, g*64+oo] = elu(go[g,n,:] + gat_b[g,:]) @ mlp_W[g] + mlp_b[g]
__global__ void k_mlp(const float* __restrict__ go, const float* __restrict__ gb,
                      const float* __restrict__ W, const float* __restrict__ b,
                      float* __restrict__ yb) {
  int t = blockIdx.x * blockDim.x + threadIdx.x;
  int g = blockIdx.z;
  if (t >= NN * NHID) return;
  int n = t >> 6, oo = t & 63;
  const float* gp = go + ((size_t)g * NN + n) * (HH * FF);
  const float* bp = gb + g * (HH * FF);
  const float* wp = W + ((size_t)g * (HH * FF)) * NHID + oo;
  float s = b[g * NHID + oo];
  #pragma unroll 8
  for (int o = 0; o < HH * FF; ++o)
    s += elu(gp[o] + bp[o]) * wp[(size_t)o * NHID];
  yb[(size_t)n * (GG * NHID) + g * NHID + oo] = s;
}

__global__ void k_out(const float* __restrict__ yb, const float* __restrict__ W,
                      const float* __restrict__ b, float* __restrict__ lg) {
  int t = blockIdx.x * blockDim.x + threadIdx.x;
  if (t >= NN * CC) return;
  int n = t / CC, c = t % CC;
  const float* y = yb + (size_t)n * (GG * NHID);
  float s = b[c];
  for (int q = 0; q < GG * NHID; ++q) s += elu(y[q]) * W[(size_t)q * CC + c];
  lg[(size_t)n * CC + c] = s;
}

__global__ void k_lsm(const float* __restrict__ lg, float* __restrict__ out) {
  int n = blockIdx.x * blockDim.x + threadIdx.x;
  if (n >= NN) return;
  float m = -1e30f;
  for (int c = 0; c < CC; ++c) m = fmaxf(m, lg[n * CC + c]);
  float s = 0.f;
  for (int c = 0; c < CC; ++c) s += __expf(lg[n * CC + c] - m);
  float l = m + __logf(s);
  for (int c = 0; c < CC; ++c) out[n * CC + c] = lg[n * CC + c] - l;
}

// ---------------------------------------------------------------------------
extern "C" void kernel_launch(void* const* d_in, const int* in_sizes, int n_in,
                              void* d_out, int out_size, void* d_ws, size_t ws_size,
                              hipStream_t stream) {
  const float* x     = (const float*)d_in[0];
  const int*   ei    = (const int*)d_in[1];
  const float* U     = (const float*)d_in[2];
  const float* psi   = (const float*)d_in[3];
  const float* gatW  = (const float*)d_in[4];
  const float* attS  = (const float*)d_in[5];
  const float* attD  = (const float*)d_in[6];
  const float* gatB  = (const float*)d_in[7];
  const float* mlpW  = (const float*)d_in[8];
  const float* mlpB  = (const float*)d_in[9];
  const float* outW  = (const float*)d_in[10];
  const float* outB  = (const float*)d_in[11];
  float* out = (float*)d_out;

  char* ws = (char*)d_ws;
  size_t off = 0;
  auto take = [&](size_t bytes) { char* p = ws + off; off += (bytes + 255) & ~(size_t)255; return p; };
  unsigned short* frag = (unsigned short*)take((size_t)KT * CT * 32 * 16 * 2); // 3.4MB
  float* ybuf  = (float*)take((size_t)NN * (JJ * FF) * 4);          // |y1| [N,96]
  float* y2buf = (float*)take((size_t)NN * (JJ * JJ * FF) * 4);     // |y2| [N,288]
  float* coef  = (float*)take((size_t)NN * (GG * FF) * 4);          // [N,416]
  float* hbuf  = (float*)take((size_t)GG * NN * (HH * FF) * 4);
  float* es    = (float*)take((size_t)GG * NN * HH * 4);
  float* ed    = (float*)take((size_t)GG * NN * HH * 4);
  unsigned* mb = (unsigned*)take((size_t)GG * NN * HH * 4);
  float* zs    = (float*)take((size_t)GG * NN * HH * 4);
  float* go    = (float*)take((size_t)GG * NN * (HH * FF) * 4);
  float* yb    = (float*)take((size_t)NN * (GG * NHID) * 4);
  float* lgts  = (float*)take((size_t)NN * CC * 4);

  // Optional 96MB cache of psi as bf16 A-fragments (written by level-1 pass,
  // read by level-2 pass). Enabled only if the workspace is big enough;
  // deterministic for a fixed harness allocation.
  const size_t psafragBytes = (size_t)JJ * RT * KT * 512 * 2;
  unsigned short* psafrag = nullptr;
  if (off + psafragBytes <= ws_size) psafrag = (unsigned short*)take(psafragBytes);

  // zero segment-max keys (0 == -inf sentinel), softmax sums, aggregation buf
  int nz1 = GG * NN * HH, nz2 = GG * NN * (HH * FF);
  k_zero<<<(nz1 + 255) / 256, 256, 0, stream>>>(mb, nz1);
  k_zero<<<(nz1 + 255) / 256, 256, 0, stream>>>((unsigned*)zs, nz1);
  k_zero<<<(nz2 + 255) / 256, 256, 0, stream>>>((unsigned*)go, nz2);

  // |x| -> fragment col-tiles 0..1
  k_pack<<<32, 256, 0, stream>>>(x, FF, 0, frag, 0, 1);
  // y1_j = psi_j @ |x|, epilogue abs -> ybuf[N, j*32+f]; spill psi bf16 frags
  k_gemm<<<dim3(32, 1, JJ), 256, 0, stream>>>(psi, (long long)NN * NN, frag, 0,
                                              nullptr, psafrag,
                                              ybuf, JJ * FF, FF, 1);
  for (int j = 0; j < JJ; ++j)  // |y1_j| -> tiles 2+2j
    k_pack<<<32, 256, 0, stream>>>(ybuf, JJ * FF, j * FF, frag, 2 + 2 * j, 0);
  // y2_{j,k} = psi_k @ |y1_j|, abs -> y2buf[N, k*96 + j*32 + f]
  k_gemm<<<dim3(32, JJ, JJ), 256, 0, stream>>>(psi, (long long)NN * NN, frag, 2,
                                               psafrag, nullptr,
                                               y2buf, JJ * JJ * FF, JJ * FF, 1);
  for (int j = 0; j < JJ; ++j)
    for (int k = 0; k < JJ; ++k)  // -> tiles 8 + (3j+k)*2 (branch order)
      k_pack<<<32, 256, 0, stream>>>(y2buf, JJ * JJ * FF, k * (JJ * FF) + j * FF,
                                     frag, 8 + (j * JJ + k) * 2, 0);
  // all 13 coefs in one U pass: coef[n, g*32+f]
  k_gemm<<<dim3(32, GG, 1), 256, 0, stream>>>(U, 0LL, frag, 0,
                                              nullptr, nullptr,
                                              coef, GG * FF, 0, 0);

  // GAT branches (z = branch)
  k_h  <<<dim3((NN * HH * FF) / 256, 1, GG), 256, 0, stream>>>(coef, gatW, gatB, hbuf);
  k_att<<<dim3((NN * HH) / 256, 1, GG), 256, 0, stream>>>(hbuf, attS, attD, es, ed);
  dim3 eg((EN * HH + 255) / 256, 1, GG);
  k_emax<<<eg, 256, 0, stream>>>(es, ed, ei, mb);
  k_esum<<<eg, 256, 0, stream>>>(es, ed, ei, mb, zs);
  k_eacc<<<eg, 256, 0, stream>>>(es, ed, ei, mb, zs, hbuf, go);
  k_mlp<<<dim3((NN * NHID) / 256, 1, GG), 256, 0, stream>>>(go, gatB, mlpW, mlpB, yb);
  k_out<<<(NN * CC + 255) / 256, 256, 0, stream>>>(yb, outW, outB, lgts);
  k_lsm<<<(NN + 255) / 256, 256, 0, stream>>>(lgts, out);
  (void)in_sizes; (void)n_in; (void)out_size;
}
